// SelfAttention_31722628448859
// MI455X (gfx1250) — matrix operand order repaired and checked by reading, not compiled, not run on hardware
//
#include <hip/hip_runtime.h>
#include <hip/hip_bf16.h>

#define D_MODEL   1024
#define N_HEADS   16
#define HEAD_DIM  64
#define SEQ       2048
#define BATCH     2
#define NROWS     (BATCH * SEQ)          // 4096

typedef __attribute__((ext_vector_type(16))) __bf16 v16bf;
typedef __attribute__((ext_vector_type(8)))  float  v8f;
typedef __attribute__((ext_vector_type(4)))  unsigned u32x4;
typedef __attribute__((ext_vector_type(8)))  int      i32x8;
typedef __attribute__((ext_vector_type(4)))  int      i32x4;

union FragU { unsigned u[8]; v16bf v; };

__device__ __forceinline__ v8f wmma_acc(v16bf a, v16bf b, v8f c) {
  return __builtin_amdgcn_wmma_f32_16x16x32_bf16(
      /*neg_a=*/false, a, /*neg_b=*/false, b,
      /*c_mod=*/(short)0, c, /*reuse_a=*/false, /*reuse_b=*/false);
}

// A-matrix 16x32 bf16 fragment (contiguous K order, matches load_fragB):
// half 0 holds K=0..15 (v_j: K=2j,2j+1), half 1 holds K=16..31.
__device__ __forceinline__ v16bf load_fragA(const __bf16* __restrict__ tile, int ld) {
  const int lane = threadIdx.x & 31;
  const int m    = lane & 15;
  const int half = lane >> 4;
  const __bf16* p = tile + (size_t)m * ld + (half << 4);
  FragU f;
#pragma unroll
  for (int j = 0; j < 8; ++j) {
    const int kk = j << 1;
    f.u[j] = *(const unsigned*)(p + kk);
  }
  return f.v;
}

// B-matrix 32x16 bf16 fragment from (N-major, K-contiguous) storage:
// half 0 holds K=0..15 (v_j: K=2j,2j+1), half 1 holds K=16..31.
__device__ __forceinline__ v16bf load_fragB(const __bf16* __restrict__ tile, int ld) {
  const int lane = threadIdx.x & 31;
  const int n    = lane & 15;
  const int half = lane >> 4;
  const __bf16* p = tile + (size_t)n * ld + (half << 4);
  FragU f;
#pragma unroll
  for (int j = 0; j < 8; ++j)
    f.u[j] = *(const unsigned*)(p + (j << 1));
  return f.v;
}

// ---- Tensor Data Mover: stage a 2D (K=32 x N=n_rows) bf16 tile into LDS ----
// D# per CDNA5 ISA: group0 {count=1, lds_addr,
// global_addr, type=2}; group1 {data_size=2B, tensor_dim0=1024, tensor_dim1=1024,
// tile_dim0=32, tile_dim1=n_rows, tensor_dim0_stride=1024}. Groups 2/3 unused (2D).
__device__ __forceinline__ void tdm_load_b_tile(const __bf16* gsrc, void* ldst,
                                                int n_rows) {
  const unsigned long long ga = (unsigned long long)(uintptr_t)gsrc;
  const unsigned lds_off = (unsigned)(uintptr_t)ldst;   // low 32b = LDS byte offset
  u32x4 g0;
  g0.x = 1u;                                            // count=1 (valid descriptor)
  g0.y = lds_off;                                       // [63:32] lds_addr
  g0.z = (unsigned)ga;                                  // [95:64] global_addr lo
  g0.w = (unsigned)((ga >> 32) & 0x1FFFFFFu) | (2u << 30); // addr[56:32] | type=2
  i32x8 g1;
  g1[0] = (int)(1u << 16);                              // data_size=1 -> 2 bytes
  g1[1] = (int)((D_MODEL & 0xFFFFu) << 16);             // tensor_dim0 lo16
  g1[2] = (int)(((unsigned)D_MODEL & 0xFFFFu) << 16);   // dim0 hi=0 | tensor_dim1 lo16
  g1[3] = (int)(32u << 16);                             // dim1 hi=0 | tile_dim0=32 (K)
  g1[4] = n_rows;                                       // tile_dim1 (N rows), tile_dim2=0
  g1[5] = D_MODEL;                                      // tensor_dim0_stride lo32
  g1[6] = 0;
  g1[7] = 0;
  i32x4 z4 = {0, 0, 0, 0};
  i32x8 z8 = {0, 0, 0, 0, 0, 0, 0, 0};
  __builtin_amdgcn_tensor_load_to_lds(g0, g1, z4, z4, z8, 0);
}

// ---------------- conversion kernels ----------------
__global__ void k_cvt_x(const float* __restrict__ in, __bf16* __restrict__ out) {
  const int stride = gridDim.x * blockDim.x;
  for (int i = blockIdx.x * blockDim.x + threadIdx.x; i < NROWS * D_MODEL; i += stride)
    out[i] = (__bf16)in[i];
}

// W stored [K_in, N_out]; emit transposed bf16 [N_out, K_in] (K contiguous).
__global__ void k_cvt_wT(const float* __restrict__ in, __bf16* __restrict__ out) {
  const int stride = gridDim.x * blockDim.x;
  for (int i = blockIdx.x * blockDim.x + threadIdx.x; i < D_MODEL * D_MODEL; i += stride) {
    const int k = i >> 10;
    const int n = i & (D_MODEL - 1);
    out[(size_t)n * D_MODEL + k] = (__bf16)in[i];
  }
}

// ---------------- blocked GEMM: [4096x1024] x [1024x1024] + bias ----------------
// Block = 256 thr = 8 waves as 2(M) x 4(N): 128x128 block tile.
// Wave = 64(M) x 32(N): 8 accumulators, 8 WMMA per K-step of 32.
// B tile (128 N x 32 K) staged in LDS by TDM, double buffered.
// MODE 0: bf16 head layout [bh, L, 64] (Q,K)   MODE 1: bf16 [bh, 64, L] (V)
// MODE 2: fp32 row-major [4096,1024]
template <int MODE>
__global__ __launch_bounds__(256) void k_gemm(const __bf16* __restrict__ A,
                                              const __bf16* __restrict__ BT,
                                              const float* __restrict__ bias,
                                              void* __restrict__ outp) {
  __shared__ __bf16 tileB[2][128][32];                  // 16 KB
  const int lane = threadIdx.x & 31;
  const int wid  = threadIdx.x >> 5;
  const int wm   = wid >> 2;                            // 0..1
  const int wn   = wid & 3;                             // 0..3
  const int bm   = blockIdx.x >> 3;                     // 32 M blocks
  const int bn   = blockIdx.x & 7;                      // 8  N blocks
  const int m0   = bm * 128 + wm * 64;
  const int n0   = bn * 128 + wn * 32;

  const __bf16* Ab  = A  + (size_t)m0 * D_MODEL;
  const __bf16* Bbl = BT + (size_t)(bn * 128) * D_MODEL;

  v8f acc[4][2] = {};

  if (wid == 0) {
    tdm_load_b_tile(Bbl, &tileB[0][0][0], 128);
    __builtin_amdgcn_s_wait_tensorcnt(0);
  }
  __syncthreads();

  int p = 0;
  for (int k = 0; k < D_MODEL; k += 32) {
    if (wid == 0 && (k + 32) < D_MODEL)
      tdm_load_b_tile(Bbl + (k + 32), &tileB[p ^ 1][0][0], 128);

    v16bf bf0 = load_fragB(&tileB[p][wn * 32 +  0][0], 32);
    v16bf bf1 = load_fragB(&tileB[p][wn * 32 + 16][0], 32);
#pragma unroll
    for (int i = 0; i < 4; ++i) {
      v16bf af = load_fragA(Ab + (size_t)(i * 16) * D_MODEL + k, D_MODEL);
      acc[i][0] = wmma_acc(af, bf0, acc[i][0]);
      acc[i][1] = wmma_acc(af, bf1, acc[i][1]);
    }

    if (wid == 0) __builtin_amdgcn_s_wait_tensorcnt(0);
    __syncthreads();
    p ^= 1;
  }

  const int n  = lane & 15;
  const int mb = (lane >> 4) << 3;
#pragma unroll
  for (int i = 0; i < 4; ++i) {
#pragma unroll
    for (int t = 0; t < 2; ++t) {
      const int col = n0 + t * 16 + n;
      const float bv = bias[col];
#pragma unroll
      for (int r = 0; r < 8; ++r) {
        const int row = m0 + i * 16 + mb + r;
        const float v = acc[i][t][r] + bv;
        if (MODE == 2) {
          ((float*)outp)[(size_t)row * D_MODEL + col] = v;
        } else {
          const int b = row >> 11, l = row & (SEQ - 1);
          const int h = col >> 6,  hd = col & (HEAD_DIM - 1);
          const int bh = b * N_HEADS + h;
          if (MODE == 0)
            ((__bf16*)outp)[((size_t)bh * SEQ + l) * HEAD_DIM + hd] = (__bf16)v;
          else
            ((__bf16*)outp)[((size_t)bh * HEAD_DIM + hd) * SEQ + l] = (__bf16)v;
        }
      }
    }
  }
}

// ---------------- flash-style attention ----------------
// grid (32 qblocks, 32 bh), 128 threads = 4 waves, each wave = 16 query rows.
__global__ __launch_bounds__(128) void k_attn(const __bf16* __restrict__ Q,
                                              const __bf16* __restrict__ K,
                                              const __bf16* __restrict__ Vt,
                                              __bf16* __restrict__ Ctx) {
  __shared__ __bf16 pbuf[4][16][64];
  const int lane = threadIdx.x & 31;
  const int wid  = threadIdx.x >> 5;
  const int bh   = blockIdx.y;
  const int b    = bh >> 4, h = bh & (N_HEADS - 1);
  const int q0   = blockIdx.x * 64 + wid * 16;

  const __bf16* Qb = Q  + ((size_t)bh * SEQ + q0) * HEAD_DIM;
  const __bf16* Kb = K  + (size_t)bh * SEQ * HEAD_DIM;
  const __bf16* Vb = Vt + (size_t)bh * HEAD_DIM * SEQ;

  const v16bf qf0 = load_fragA(Qb + 0,  HEAD_DIM);
  const v16bf qf1 = load_fragA(Qb + 32, HEAD_DIM);

  v8f o0 = {}, o1 = {}, o2 = {}, o3 = {};
  float rm[8], rl[8];
#pragma unroll
  for (int r = 0; r < 8; ++r) { rm[r] = -3.0e38f; rl[r] = 0.0f; }

  for (int kb = 0; kb < SEQ; kb += 64) {
    v8f s0 = {}, s1 = {}, s2 = {}, s3 = {};
    {
      v16bf kf;
      kf = load_fragB(Kb + (size_t)(kb +  0) * HEAD_DIM + 0,  HEAD_DIM); s0 = wmma_acc(qf0, kf, s0);
      kf = load_fragB(Kb + (size_t)(kb +  0) * HEAD_DIM + 32, HEAD_DIM); s0 = wmma_acc(qf1, kf, s0);
      kf = load_fragB(Kb + (size_t)(kb + 16) * HEAD_DIM + 0,  HEAD_DIM); s1 = wmma_acc(qf0, kf, s1);
      kf = load_fragB(Kb + (size_t)(kb + 16) * HEAD_DIM + 32, HEAD_DIM); s1 = wmma_acc(qf1, kf, s1);
      kf = load_fragB(Kb + (size_t)(kb + 32) * HEAD_DIM + 0,  HEAD_DIM); s2 = wmma_acc(qf0, kf, s2);
      kf = load_fragB(Kb + (size_t)(kb + 32) * HEAD_DIM + 32, HEAD_DIM); s2 = wmma_acc(qf1, kf, s2);
      kf = load_fragB(Kb + (size_t)(kb + 48) * HEAD_DIM + 0,  HEAD_DIM); s3 = wmma_acc(qf0, kf, s3);
      kf = load_fragB(Kb + (size_t)(kb + 48) * HEAD_DIM + 32, HEAD_DIM); s3 = wmma_acc(qf1, kf, s3);
    }
#pragma unroll
    for (int r = 0; r < 8; ++r) {
      const float sc = 0.125f;                  // 1/sqrt(64)
      s0[r] *= sc; s1[r] *= sc; s2[r] *= sc; s3[r] *= sc;
      float t = fmaxf(fmaxf(s0[r], s1[r]), fmaxf(s2[r], s3[r]));
#pragma unroll
      for (int m = 8; m >= 1; m >>= 1) t = fmaxf(t, __shfl_xor(t, m, 32));
      const float mn = fmaxf(rm[r], t);
      const float cf = __expf(rm[r] - mn);
      rm[r] = mn;
      o0[r] *= cf; o1[r] *= cf; o2[r] *= cf; o3[r] *= cf;
      s0[r] = __expf(s0[r] - mn);
      s1[r] = __expf(s1[r] - mn);
      s2[r] = __expf(s2[r] - mn);
      s3[r] = __expf(s3[r] - mn);
      float rs = s0[r] + s1[r] + s2[r] + s3[r];
#pragma unroll
      for (int m = 8; m >= 1; m >>= 1) rs += __shfl_xor(rs, m, 32);
      rl[r] = rl[r] * cf + rs;
    }
    __syncthreads();
    {
      const int colc = lane & 15;
      const int mb   = (lane >> 4) << 3;
#pragma unroll
      for (int r = 0; r < 8; ++r) {
        pbuf[wid][mb + r][ 0 + colc] = (__bf16)s0[r];
        pbuf[wid][mb + r][16 + colc] = (__bf16)s1[r];
        pbuf[wid][mb + r][32 + colc] = (__bf16)s2[r];
        pbuf[wid][mb + r][48 + colc] = (__bf16)s3[r];
      }
    }
    __syncthreads();
    const v16bf pf0 = load_fragA(&pbuf[wid][0][0],  64);
    const v16bf pf1 = load_fragA(&pbuf[wid][0][32], 64);
    {
      v16bf vf;
      vf = load_fragB(Vb + (size_t) 0 * SEQ + kb +  0, SEQ); o0 = wmma_acc(pf0, vf, o0);
      vf = load_fragB(Vb + (size_t) 0 * SEQ + kb + 32, SEQ); o0 = wmma_acc(pf1, vf, o0);
      vf = load_fragB(Vb + (size_t)16 * SEQ + kb +  0, SEQ); o1 = wmma_acc(pf0, vf, o1);
      vf = load_fragB(Vb + (size_t)16 * SEQ + kb + 32, SEQ); o1 = wmma_acc(pf1, vf, o1);
      vf = load_fragB(Vb + (size_t)32 * SEQ + kb +  0, SEQ); o2 = wmma_acc(pf0, vf, o2);
      vf = load_fragB(Vb + (size_t)32 * SEQ + kb + 32, SEQ); o2 = wmma_acc(pf1, vf, o2);
      vf = load_fragB(Vb + (size_t)48 * SEQ + kb +  0, SEQ); o3 = wmma_acc(pf0, vf, o3);
      vf = load_fragB(Vb + (size_t)48 * SEQ + kb + 32, SEQ); o3 = wmma_acc(pf1, vf, o3);
    }
  }

  const int n  = lane & 15;
  const int mb = (lane >> 4) << 3;
#pragma unroll
  for (int r = 0; r < 8; ++r) {
    const int l = q0 + mb + r;
    const float inv = 1.0f / rl[r];
    __bf16* dst = Ctx + ((size_t)(b * SEQ + l)) * D_MODEL + h * HEAD_DIM;
    dst[ 0 + n] = (__bf16)(o0[r] * inv);
    dst[16 + n] = (__bf16)(o1[r] * inv);
    dst[32 + n] = (__bf16)(o2[r] * inv);
    dst[48 + n] = (__bf16)(o3[r] * inv);
  }
}

// ---------------- launcher ----------------
extern "C" void kernel_launch(void* const* d_in, const int* in_sizes, int n_in,
                              void* d_out, int out_size, void* d_ws, size_t ws_size,
                              hipStream_t stream) {
  const float* x  = (const float*)d_in[0];
  const float* Wq = (const float*)d_in[1];
  const float* bq = (const float*)d_in[2];
  const float* Wk = (const float*)d_in[3];
  const float* bk = (const float*)d_in[4];
  const float* Wv = (const float*)d_in[5];
  const float* bv = (const float*)d_in[6];
  const float* Wo = (const float*)d_in[7];
  const float* bo = (const float*)d_in[8];

  char* ws = (char*)d_ws;
  constexpr size_t SZ_X = (size_t)NROWS * D_MODEL * 2;      // 8 MB
  constexpr size_t SZ_W = (size_t)D_MODEL * D_MODEL * 2;    // 2 MB
  constexpr size_t SZ_H = (size_t)NROWS * D_MODEL * 2;      // 8 MB
  __bf16* Xbf = (__bf16*)(ws);
  __bf16* WqT = (__bf16*)(ws + SZ_X);
  __bf16* WkT = (__bf16*)(ws + SZ_X + SZ_W);
  __bf16* WvT = (__bf16*)(ws + SZ_X + 2 * SZ_W);
  __bf16* WoT = (__bf16*)(ws + SZ_X + 3 * SZ_W);
  __bf16* Qh  = (__bf16*)(ws + SZ_X + 4 * SZ_W);
  __bf16* Kh  = (__bf16*)(ws + SZ_X + 4 * SZ_W + SZ_H);
  __bf16* Vt  = (__bf16*)(ws + SZ_X + 4 * SZ_W + 2 * SZ_H);
  __bf16* Ctx = (__bf16*)(ws + SZ_X + 4 * SZ_W + 3 * SZ_H);

  k_cvt_x <<<2048, 256, 0, stream>>>(x, Xbf);
  k_cvt_wT<<<1024, 256, 0, stream>>>(Wq, WqT);
  k_cvt_wT<<<1024, 256, 0, stream>>>(Wk, WkT);
  k_cvt_wT<<<1024, 256, 0, stream>>>(Wv, WvT);
  k_cvt_wT<<<1024, 256, 0, stream>>>(Wo, WoT);

  k_gemm<0><<<256, 256, 0, stream>>>(Xbf, WqT, bq, (void*)Qh);
  k_gemm<0><<<256, 256, 0, stream>>>(Xbf, WkT, bk, (void*)Kh);
  k_gemm<1><<<256, 256, 0, stream>>>(Xbf, WvT, bv, (void*)Vt);

  k_attn<<<dim3(32, 32), 128, 0, stream>>>(Qh, Kh, Vt, Ctx);

  k_gemm<2><<<256, 256, 0, stream>>>(Ctx, WoT, bo, d_out);
}
